// FlaxDonutSwinLayer_6390911336996
// MI455X (gfx1250) — matrix operand order repaired and checked by reading, not compiled, not run on hardware
//
#include <hip/hip_runtime.h>
#include <math.h>

typedef __bf16 bf16;
typedef __attribute__((ext_vector_type(16))) __bf16 v16bf;
typedef __attribute__((ext_vector_type(8)))  __bf16 v8bf;
typedef __attribute__((ext_vector_type(8)))  float  v8f;

// ---------------------------------------------------------------------------
// WMMA helper: D = A(16x32 bf16) * B(32x16 bf16) + C(16x16 f32)
// ---------------------------------------------------------------------------
__device__ __forceinline__ v8f wmma_bf16(v16bf a, v16bf b, v8f c) {
  return __builtin_amdgcn_wmma_f32_16x16x32_bf16(false, a, false, b, (short)0, c,
                                                 false, false);
}

// A fragment from a row-major bf16 matrix in LDS.
// Lane L: M = m0 + (L&15); K = k0 + 16*(L>>4) + e  (16 contiguous elements).
__device__ __forceinline__ v16bf load_frag_A(const bf16* base, int rowStride,
                                             int m0, int k0, int lane) {
  const bf16* r = base + (m0 + (lane & 15)) * rowStride + k0 + 16 * (lane >> 4);
  v8bf lo = *(const v8bf*)(r);
  v8bf hi = *(const v8bf*)(r + 8);
  return __builtin_shufflevector(lo, hi, 0, 1, 2, 3, 4, 5, 6, 7,
                                 8, 9, 10, 11, 12, 13, 14, 15);
}

// B fragment where B[k][n] = base[(n0 + n)*rowStride + c0 + k]  (e.g. Q*K^T, P*V).
// Lane L: N = n0 + (L&15); K = c0 + 16*(L>>4) + e  (16 contiguous elements).
__device__ __forceinline__ v16bf load_frag_Brow(const bf16* base, int rowStride,
                                                int n0, int c0, int lane) {
  const bf16* r = base + (n0 + (lane & 15)) * rowStride + c0 + 16 * (lane >> 4);
  v8bf lo = *(const v8bf*)(r);
  v8bf hi = *(const v8bf*)(r + 8);
  return __builtin_shufflevector(lo, hi, 0, 1, 2, 3, 4, 5, 6, 7,
                                 8, 9, 10, 11, 12, 13, 14, 15);
}

// B fragment from pre-packed global weights: one 32-byte aligned load per lane.
__device__ __forceinline__ v16bf load_frag_Bpk(const bf16* pack, int fragIdx,
                                               int lane) {
  return *(const v16bf*)(pack + (((long)fragIdx * 32 + lane) << 4));
}

// ---------------------------------------------------------------------------
// Weight packing: fp32 row-major W[K][N] -> bf16 WMMA-B fragments.
// Global frag ids: q_w [0,32) k_w [32,64) v_w [64,96) proj_w [96,128)
//                  fc1 [128,256)  fc2 [256,384)
// ---------------------------------------------------------------------------
__global__ __launch_bounds__(256) void pack_weights_kernel(
    const float* __restrict__ qw, const float* __restrict__ kw,
    const float* __restrict__ vw, const float* __restrict__ pw,
    const float* __restrict__ f1, const float* __restrict__ f2,
    bf16* __restrict__ pack) {
  int gid = blockIdx.x * 256 + threadIdx.x;  // 384 frags * 32 lanes = 12288
  int f = gid >> 5, lane = gid & 31;
  const float* W;
  int Nd, kt, nt;
  if (f < 128) {
    int mat = f >> 5, fl = f & 31;
    W = (mat == 0) ? qw : (mat == 1) ? kw : (mat == 2) ? vw : pw;
    Nd = 128; nt = fl >> 2; kt = fl & 3;
  } else if (f < 256) {
    int fl = f - 128;
    W = f1; Nd = 512; nt = fl >> 2; kt = fl & 3;
  } else {
    int fl = f - 256;
    W = f2; Nd = 128; nt = fl >> 4; kt = fl & 15;
  }
  bf16* dst = pack + ((long)f * 512 + lane * 16);
  int kbase = kt * 32 + 16 * (lane >> 4);
  int n = nt * 16 + (lane & 15);
#pragma unroll
  for (int e = 0; e < 16; ++e) dst[e] = (bf16)W[(kbase + e) * Nd + n];
}

// ---------------------------------------------------------------------------
// Fused shifted-window attention: LN1 + shift + QKV + attn + proj + residual.
// One block per window (4096 blocks), 256 threads = 8 waves.
// ---------------------------------------------------------------------------
#define XLN_OFF 0          // 64 x 136 bf16 = 17408 B
#define Q_OFF   17408      // 64 x 136 bf16
#define K_OFF   34816      // 64 x 136 bf16
#define VT_OFF  52224      // 128 x 72 bf16 (V transposed: [ch][tok])
#define S_OFF   70656      // 4 x 64 x 66 f32 = 67584 B
#define P_OFF   138240     // 4 x 64 x 72 bf16 = 36864 B
#define O_OFF   175104     // 64 x 136 bf16
#define RPB_OFF 192512     // 225 x 4 f32 = 3600 B
#define ATTN_SMEM 196112

__global__ __launch_bounds__(256) void swin_attn_kernel(
    const float* __restrict__ hs, const float* __restrict__ ln1_s,
    const float* __restrict__ ln1_b, const bf16* __restrict__ pack,
    const float* __restrict__ q_bias, const float* __restrict__ k_bias,
    const float* __restrict__ v_bias, const float* __restrict__ proj_bias,
    const float* __restrict__ rpb, float* __restrict__ out) {
  extern __shared__ char smem[];
  bf16*  xln  = (bf16*)(smem + XLN_OFF);
  bf16*  qs   = (bf16*)(smem + Q_OFF);
  bf16*  ks   = (bf16*)(smem + K_OFF);
  bf16*  vt   = (bf16*)(smem + VT_OFF);
  float* sS   = (float*)(smem + S_OFF);
  bf16*  pS   = (bf16*)(smem + P_OFF);
  bf16*  oS   = (bf16*)(smem + O_OFF);
  float* rpbl = (float*)(smem + RPB_OFF);

  const int tid = threadIdx.x;
  const int lane = tid & 31;
  // wave id, forced into an SGPR so all tile indices are compiler-visibly uniform
  const int wvu = __builtin_amdgcn_readfirstlane(tid >> 5);
  const int w = blockIdx.x;           // 0..4095
  const int b = w >> 10;              // batch
  const int wimg = w & 1023;
  const int wi = wimg >> 5, wj = wimg & 31;

  // warm L2/WGP$ for the packed-weight B-fragment stream (global_prefetch_b8)
  __builtin_prefetch((const void*)(pack + (long)wvu * 8192), 0, 1);

  // stage relative-position-bias table into LDS
  for (int i = tid; i < 900; i += 256) rpbl[i] = rpb[i];

  // ---- stage 0: gather (cyclic shift + window partition) + LayerNorm1 ----
  {
    const int tok = tid >> 2, q4 = tid & 3;
    const int rr = tok >> 3, cc = tok & 7;
    const int hp = (wi * 8 + rr + 4) & 255;
    const int wp = (wj * 8 + cc + 4) & 255;
    const long srcBase = ((long)(b << 16) + hp * 256 + wp) * 128;
    const float* src = hs + srcBase + q4 * 32;
    float xr[32];
    float s1 = 0.f, s2 = 0.f;
#pragma unroll
    for (int i = 0; i < 32; ++i) {
      float v = src[i];
      xr[i] = v; s1 += v; s2 += v * v;
    }
    s1 += __shfl_xor(s1, 1); s2 += __shfl_xor(s2, 1);
    s1 += __shfl_xor(s1, 2); s2 += __shfl_xor(s2, 2);
    float mean = s1 * (1.f / 128.f);
    float var = s2 * (1.f / 128.f) - mean * mean;
    float rstd = rsqrtf(var + 1e-5f);
#pragma unroll
    for (int i = 0; i < 32; ++i) {
      int ch = q4 * 32 + i;
      float v = (xr[i] - mean) * rstd * ln1_s[ch] + ln1_b[ch];
      xln[tok * 136 + ch] = (bf16)v;
    }
  }
  __syncthreads();

  // ---- stage 1: QKV GEMMs. Per wave: mt = wvu>>1, 4 consecutive n-tiles. ----
  // kt-outer register blocking: one A fragment feeds 4 independent WMMAs.
  const float qscale = 0.17677669529663687f;  // 1/sqrt(32)
  {
    const int mt = wvu >> 1;
    const int ntBase = (wvu & 1) * 4;
    for (int mat = 0; mat < 3; ++mat) {           // uniform: 0=Q 1=K 2=V
      v8f acc[4] = {{}, {}, {}, {}};
#pragma unroll
      for (int kt = 0; kt < 4; ++kt) {
        v16bf a = load_frag_A(xln, 136, mt * 16, kt * 32, lane);
#pragma unroll
        for (int j = 0; j < 4; ++j) {
          v16bf bm = load_frag_Bpk(pack, mat * 32 + (ntBase + j) * 4 + kt, lane);
          acc[j] = wmma_bf16(a, bm, acc[j]);
        }
      }
      const float* bp = (mat == 0) ? q_bias : (mat == 1) ? k_bias : v_bias;
#pragma unroll
      for (int j = 0; j < 4; ++j) {
        int n = (ntBase + j) * 16 + (lane & 15);
        float bv = bp[n];
#pragma unroll
        for (int e = 0; e < 8; ++e) {
          int m = mt * 16 + e + 8 * (lane >> 4);
          float v = acc[j][e] + bv;
          if (mat == 0)      qs[m * 136 + n] = (bf16)(v * qscale);
          else if (mat == 1) ks[m * 136 + n] = (bf16)v;
          else               vt[n * 72 + m] = (bf16)v;   // transposed
        }
      }
    }
  }
  __syncthreads();

  // ---- stage 2: S = Q * K^T. Per wave: head h = wvu>>1, 2 m-tiles x 4 n-tiles.
  {
    const int h = wvu >> 1;
    const int mtBase = (wvu & 1) * 2;
    v16bf bk[4];
#pragma unroll
    for (int nt = 0; nt < 4; ++nt)
      bk[nt] = load_frag_Brow(ks, 136, nt * 16, h * 32, lane);
#pragma unroll
    for (int mi = 0; mi < 2; ++mi) {
      int mt = mtBase + mi;
      v16bf a = load_frag_A(qs, 136, mt * 16, h * 32, lane);
#pragma unroll
      for (int nt = 0; nt < 4; ++nt) {
        v8f acc = {};
        acc = wmma_bf16(a, bk[nt], acc);
        int n = nt * 16 + (lane & 15);
#pragma unroll
        for (int e = 0; e < 8; ++e) {
          int m = mt * 16 + e + 8 * (lane >> 4);
          sS[(h * 64 + m) * 66 + n] = acc[e];
        }
      }
    }
  }
  __syncthreads();

  // ---- stage 3: rel-pos bias + shift mask + softmax (one thread per (h,q)) ----
  {
    const int h = wvu >> 1;          // == tid>>6, wave-uniform
    const int qq = tid & 63;
    float* row = sS + (h * 64 + qq) * 66;
    int qr = qq >> 3, qc = qq & 7;
    int catq = ((wi == 31) ? (qr < 4 ? 1 : 2) : 0) * 3 +
               ((wj == 31) ? (qc < 4 ? 1 : 2) : 0);
    float mx = -1e30f;
#pragma unroll
    for (int i = 0; i < 64; ++i) {
      int kr = i >> 3, kc = i & 7;
      int ridx = (qr - kr + 7) * 15 + (qc - kc + 7);
      int catk = ((wi == 31) ? (kr < 4 ? 1 : 2) : 0) * 3 +
                 ((wj == 31) ? (kc < 4 ? 1 : 2) : 0);
      float v = row[i] + rpbl[ridx * 4 + h] + ((catq != catk) ? -100.f : 0.f);
      row[i] = v;
      mx = fmaxf(mx, v);
    }
    float sum = 0.f;
#pragma unroll
    for (int i = 0; i < 64; ++i) {
      float e = __expf(row[i] - mx);
      row[i] = e; sum += e;
    }
    float inv = 1.f / sum;
    bf16* prow = pS + (h * 64 + qq) * 72;
#pragma unroll
    for (int i = 0; i < 64; ++i) prow[i] = (bf16)(row[i] * inv);
  }
  __syncthreads();

  // ---- stage 4: O = P * V. Per wave: head h = wvu>>1, 2 m-tiles x 2 n-tiles. --
  {
    const int h = wvu >> 1;
    const int mtBase = (wvu & 1) * 2;
    v16bf bv4[2][2];
#pragma unroll
    for (int nt = 0; nt < 2; ++nt)
#pragma unroll
      for (int kt = 0; kt < 2; ++kt)
        bv4[nt][kt] = load_frag_Brow(vt, 72, h * 32 + nt * 16, kt * 32, lane);
#pragma unroll
    for (int mi = 0; mi < 2; ++mi) {
      int mt = mtBase + mi;
      v8f acc[2] = {{}, {}};
#pragma unroll
      for (int kt = 0; kt < 2; ++kt) {
        v16bf a = load_frag_A(pS + h * 64 * 72, 72, mt * 16, kt * 32, lane);
#pragma unroll
        for (int nt = 0; nt < 2; ++nt)
          acc[nt] = wmma_bf16(a, bv4[nt][kt], acc[nt]);
      }
#pragma unroll
      for (int nt = 0; nt < 2; ++nt) {
        int n = h * 32 + nt * 16 + (lane & 15);
#pragma unroll
        for (int e = 0; e < 8; ++e) {
          int m = mt * 16 + e + 8 * (lane >> 4);
          oS[m * 136 + n] = (bf16)acc[nt][e];
        }
      }
    }
  }
  __syncthreads();

  // ---- stage 5: proj GEMM + residual + inverse shift scatter ----
  {
    const int mt = wvu >> 1;
    const int ntBase = (wvu & 1) * 4;
    v8f acc[4] = {{}, {}, {}, {}};
#pragma unroll
    for (int kt = 0; kt < 4; ++kt) {
      v16bf a = load_frag_A(oS, 136, mt * 16, kt * 32, lane);
#pragma unroll
      for (int j = 0; j < 4; ++j) {
        v16bf bm = load_frag_Bpk(pack, 96 + (ntBase + j) * 4 + kt, lane);
        acc[j] = wmma_bf16(a, bm, acc[j]);
      }
    }
#pragma unroll
    for (int j = 0; j < 4; ++j) {
      int n = (ntBase + j) * 16 + (lane & 15);
      float bv = proj_bias[n];
#pragma unroll
      for (int e = 0; e < 8; ++e) {
        int m = mt * 16 + e + 8 * (lane >> 4);
        int mr = m >> 3, mc = m & 7;
        int hp = (wi * 8 + mr + 4) & 255;
        int wp = (wj * 8 + mc + 4) & 255;
        long o = ((long)(b << 16) + hp * 256 + wp) * 128 + n;
        out[o] = hs[o] + acc[j][e] + bv;   // shortcut + attention output
      }
    }
  }
}

// ---------------------------------------------------------------------------
// Fused MLP: LN2 + FC1 + exact GELU + FC2 + residual (in-place on d_out).
// One block per 64-token row tile (4096 blocks), hidden 64x512 kept in LDS.
// ---------------------------------------------------------------------------
#define MLP_A_OFF 0        // 64 x 136 bf16 = 17408 B
#define MLP_H_OFF 17408    // 64 x 520 bf16 = 66560 B
#define MLP_SMEM  83968

__global__ __launch_bounds__(256) void swin_mlp_kernel(
    const float* __restrict__ ln2_s, const float* __restrict__ ln2_b,
    const bf16* __restrict__ pack, const float* __restrict__ fc1_b,
    const float* __restrict__ fc2_b, float* __restrict__ out) {
  extern __shared__ char smem[];
  bf16* aT = (bf16*)(smem + MLP_A_OFF);
  bf16* hT = (bf16*)(smem + MLP_H_OFF);

  const int tid = threadIdx.x;
  const int lane = tid & 31;
  const int wvu = __builtin_amdgcn_readfirstlane(tid >> 5);
  const long rowBase = (long)blockIdx.x * 64;

  __builtin_prefetch((const void*)(pack + 65536 + (long)wvu * 16384), 0, 1);

  // ---- LN2 into bf16 tile ----
  {
    const int tok = tid >> 2, q4 = tid & 3;
    const float* src = out + (rowBase + tok) * 128 + q4 * 32;
    float xr[32];
    float s1 = 0.f, s2 = 0.f;
#pragma unroll
    for (int i = 0; i < 32; ++i) {
      float v = src[i];
      xr[i] = v; s1 += v; s2 += v * v;
    }
    s1 += __shfl_xor(s1, 1); s2 += __shfl_xor(s2, 1);
    s1 += __shfl_xor(s1, 2); s2 += __shfl_xor(s2, 2);
    float mean = s1 * (1.f / 128.f);
    float var = s2 * (1.f / 128.f) - mean * mean;
    float rstd = rsqrtf(var + 1e-5f);
#pragma unroll
    for (int i = 0; i < 32; ++i) {
      int ch = q4 * 32 + i;
      float v = (xr[i] - mean) * rstd * ln2_s[ch] + ln2_b[ch];
      aT[tok * 136 + ch] = (bf16)v;
    }
  }
  __syncthreads();

  // ---- FC1 (M=64,N=512,K=128) + exact GELU. Per wave: mt=wvu>>1, 16 n-tiles
  //      in 4 register-blocked groups of 4. ----
  {
    const int mt = wvu >> 1;
    const int ntBase = (wvu & 1) * 16;
    for (int g = 0; g < 4; ++g) {
      v8f acc[4] = {{}, {}, {}, {}};
#pragma unroll
      for (int kt = 0; kt < 4; ++kt) {
        v16bf a = load_frag_A(aT, 136, mt * 16, kt * 32, lane);
#pragma unroll
        for (int j = 0; j < 4; ++j) {
          int nt = ntBase + g * 4 + j;
          v16bf bm = load_frag_Bpk(pack, 128 + nt * 4 + kt, lane);
          acc[j] = wmma_bf16(a, bm, acc[j]);
        }
      }
#pragma unroll
      for (int j = 0; j < 4; ++j) {
        int n = (ntBase + g * 4 + j) * 16 + (lane & 15);
        float bv = fc1_b[n];
#pragma unroll
        for (int e = 0; e < 8; ++e) {
          int m = mt * 16 + e + 8 * (lane >> 4);
          float v = acc[j][e] + bv;
          v = 0.5f * v * (1.f + erff(v * 0.70710678118654752f));
          hT[m * 520 + n] = (bf16)v;
        }
      }
    }
  }
  __syncthreads();

  // ---- FC2 (M=64,N=128,K=512) + residual, in place. kt-outer blocking. ----
  {
    const int mt = wvu >> 1;
    const int ntBase = (wvu & 1) * 4;
    v8f acc[4] = {{}, {}, {}, {}};
#pragma unroll
    for (int kt = 0; kt < 16; ++kt) {
      v16bf a = load_frag_A(hT, 520, mt * 16, kt * 32, lane);
#pragma unroll
      for (int j = 0; j < 4; ++j) {
        v16bf bm = load_frag_Bpk(pack, 256 + (ntBase + j) * 16 + kt, lane);
        acc[j] = wmma_bf16(a, bm, acc[j]);
      }
    }
#pragma unroll
    for (int j = 0; j < 4; ++j) {
      int n = (ntBase + j) * 16 + (lane & 15);
      float bv = fc2_b[n];
#pragma unroll
      for (int e = 0; e < 8; ++e) {
        int m = mt * 16 + e + 8 * (lane >> 4);
        long o = (rowBase + m) * 128 + n;
        out[o] = out[o] + acc[j][e] + bv;
      }
    }
  }
}

// ---------------------------------------------------------------------------
extern "C" void kernel_launch(void* const* d_in, const int* in_sizes, int n_in,
                              void* d_out, int out_size, void* d_ws,
                              size_t ws_size, hipStream_t stream) {
  (void)in_sizes; (void)n_in; (void)out_size; (void)ws_size;
  const float* hs     = (const float*)d_in[0];
  const float* ln1_s  = (const float*)d_in[1];
  const float* ln1_b  = (const float*)d_in[2];
  const float* q_w    = (const float*)d_in[3];
  const float* q_b    = (const float*)d_in[4];
  const float* k_w    = (const float*)d_in[5];
  const float* k_b    = (const float*)d_in[6];
  const float* v_w    = (const float*)d_in[7];
  const float* v_b    = (const float*)d_in[8];
  const float* p_w    = (const float*)d_in[9];
  const float* p_b    = (const float*)d_in[10];
  const float* rpb    = (const float*)d_in[11];
  const float* ln2_s  = (const float*)d_in[12];
  const float* ln2_b  = (const float*)d_in[13];
  const float* fc1_w  = (const float*)d_in[14];
  const float* fc1_b  = (const float*)d_in[15];
  const float* fc2_w  = (const float*)d_in[16];
  const float* fc2_b  = (const float*)d_in[17];
  float* out = (float*)d_out;
  bf16* pack = (bf16*)d_ws;  // 196608 bf16 = 384 KB of packed weights

  pack_weights_kernel<<<48, 256, 0, stream>>>(q_w, k_w, v_w, p_w, fc1_w, fc2_w,
                                              pack);
  swin_attn_kernel<<<4096, 256, ATTN_SMEM, stream>>>(
      hs, ln1_s, ln1_b, pack, q_b, k_b, v_b, p_b, rpb, out);
  swin_mlp_kernel<<<4096, 256, MLP_SMEM, stream>>>(ln2_s, ln2_b, pack, fc1_b,
                                                   fc2_b, out);
}
